// GINLayer_44882408243752
// MI455X (gfx1250) — matrix operand order repaired and checked by reading, not compiled, not run on hardware
//
#include <hip/hip_runtime.h>
#include <hip/hip_bf16.h>
#include <stdint.h>

#define N_NODES 10000
#define N_EDGES 160000
#define D 512

typedef __attribute__((ext_vector_type(16))) __bf16 v16bf;
typedef __attribute__((ext_vector_type(8)))  __bf16 v8bf;
typedef __attribute__((ext_vector_type(8)))  float  v8f;
typedef __attribute__((ext_vector_type(4)))  unsigned int v4u;
typedef __attribute__((ext_vector_type(8)))  int v8i;
typedef __attribute__((ext_vector_type(4)))  int v4i;

// LDS A panel: 16 rows x 512 bf16, row stride 520 elems (1040 B = 260 dwords)
// -> lanes 0..15 reading different rows hit different banks (260 % 64 != 0).
#define LDS_ROW 520

__device__ __forceinline__ v16bf ld16g(const __bf16* p) {
  v8bf lo = *(const v8bf*)p;
  v8bf hi = *(const v8bf*)(p + 8);
  return __builtin_shufflevector(lo, hi, 0, 1, 2, 3, 4, 5, 6, 7,
                                 8, 9, 10, 11, 12, 13, 14, 15);
}

// ---------------------------------------------------------------------------
// 1) agg = h   (fuses the "x = h + agg" add by seeding the accumulator)
// ---------------------------------------------------------------------------
__global__ __launch_bounds__(256) void init_agg_kernel(const float4* __restrict__ h,
                                                       float4* __restrict__ agg, int n4) {
  int i = blockIdx.x * blockDim.x + threadIdx.x;
  if (i < n4) agg[i] = h[i];
}

// ---------------------------------------------------------------------------
// 2) scatter-add: one thread per (edge, 4-float chunk). h/agg are L2-resident
//    (20.5 MB each vs 192 MB L2), so random gather + fp32 atomics stay on-chip.
// ---------------------------------------------------------------------------
__global__ __launch_bounds__(256) void scatter_kernel(const float* __restrict__ h,
                                                      const long long* __restrict__ ei,
                                                      float* __restrict__ agg) {
  int gid = blockIdx.x * blockDim.x + threadIdx.x;   // 160000*128 = 20,480,000
  int e = gid >> 7;
  int c = (gid & 127) << 2;
  if (e >= N_EDGES) return;
  int s = (int)ei[e];             // send
  int r = (int)ei[N_EDGES + e];   // rec
  float4 v = *(const float4*)(h + (size_t)s * D + c);
  float* dst = agg + (size_t)r * D + c;
  unsafeAtomicAdd(dst + 0, v.x);
  unsafeAtomicAdd(dst + 1, v.y);
  unsafeAtomicAdd(dst + 2, v.z);
  unsafeAtomicAdd(dst + 3, v.w);
}

// ---------------------------------------------------------------------------
// 3) fp32 -> bf16 row-major convert
// ---------------------------------------------------------------------------
__global__ __launch_bounds__(256) void cvt_kernel(const float* __restrict__ src,
                                                  __bf16* __restrict__ dst, int n) {
  int i = (blockIdx.x * blockDim.x + threadIdx.x) * 4;
  if (i < n) {
    float4 v = *(const float4*)(src + i);
    dst[i + 0] = (__bf16)v.x;
    dst[i + 1] = (__bf16)v.y;
    dst[i + 2] = (__bf16)v.z;
    dst[i + 3] = (__bf16)v.w;
  }
}

// ---------------------------------------------------------------------------
// 4) Repack W (K=512 x N=512, row-major fp32) into the WMMA B-operand layout
//    (bf16). Contiguous k order (this code's, not the ISA table): B 32x16 tile, lane<16 holds column n with K=0..15
//    (2 per VGPR), lane>=16 holds column n with K=16..31.
//    Wpack index = ((n_tile*16 + k_tile)*32 + lane)*16 + j  (contiguous/lane)
// ---------------------------------------------------------------------------
__global__ __launch_bounds__(256) void packW_kernel(const float* __restrict__ W,
                                                    __bf16* __restrict__ Wp) {
  int t = blockIdx.x * blockDim.x + threadIdx.x;  // 32 n_tiles * 16 k_tiles * 32 lanes
  int lane   = t & 31;
  int k_tile = (t >> 5) & 15;
  int n_tile = t >> 9;
  int n     = n_tile * 16 + (lane & 15);
  int kbase = k_tile * 32 + ((lane >> 4) ? 16 : 0);
  __bf16* out = Wp + (size_t)t * 16;
#pragma unroll
  for (int j = 0; j < 16; ++j)
    out[j] = (__bf16)W[(size_t)(kbase + j) * D + n];
}

// ---------------------------------------------------------------------------
// TDM: DMA one 16x512 bf16 A panel (row-major, tile == tensor) into LDS with
// 16 B of padding appended per 1024 B row (pad_interval=256 dw, pad_amount=4 dw)
// so the LDS row stride is 1040 B. D# per ISA 8.3-8.5; workgroup_mask=0
// (not in a cluster); data_size=1 (2-byte elements); type=2 ("image").
// ---------------------------------------------------------------------------
#if __has_builtin(__builtin_amdgcn_tensor_load_to_lds)
#define HAVE_TDM 1
__device__ __forceinline__ void tdm_load_a_panel(const __bf16* gptr, unsigned lds_off) {
  unsigned long long ga = (unsigned long long)(uintptr_t)gptr;
  v4u g0;
  g0[0] = 1u;                                        // count=1, user mode
  g0[1] = lds_off;                                   // lds_addr (bytes)
  g0[2] = (unsigned)(ga & 0xFFFFFFFFu);              // global_addr[31:0]
  g0[3] = (unsigned)((ga >> 32) & 0x1FFFFFFu)        // global_addr[56:32]
          | (2u << 30);                              // type = 2
  v8i g1;
  g1[0] = (1 << 16)      // data_size = 2 bytes
        | (1 << 20)      // pad_enable
        | (7 << 22)      // pad_interval: 256 dwords
        | (3 << 25);     // pad_amount: 4 dwords
  g1[1] = (512 << 16);   // tensor_dim0 = 512 (low 16 bits)
  g1[2] = (16 << 16);    // tensor_dim0 hi = 0 ; tensor_dim1 = 16
  g1[3] = (512 << 16);   // tensor_dim1 hi = 0 ; tile_dim0 = 512
  g1[4] = 16;            // tile_dim1 = 16 ; tile_dim2 = 0
  g1[5] = 512;           // tensor_dim0_stride = 512 elements (low 32)
  g1[6] = 0;
  g1[7] = 0;
  v4i z4 = {0, 0, 0, 0};
#if defined(__clang_major__) && (__clang_major__ >= 23)
  v8i z8 = {0, 0, 0, 0, 0, 0, 0, 0};
  __builtin_amdgcn_tensor_load_to_lds(g0, g1, z4, z4, z8, 0);
#else
  __builtin_amdgcn_tensor_load_to_lds(g0, g1, z4, z4, 0);
#endif
}
#else
#define HAVE_TDM 0
#endif

// ---------------------------------------------------------------------------
// 5) GEMM: out[M=10000, N=512] = act(X @ W + bias)
//    Workgroup = one 16-row M tile staged once into LDS via TDM; wave w covers
//    columns [w*64, w*64+64) as 4 fp32 accumulators; K loop = 16 steps of
//    v_wmma_f32_16x16x32_bf16 with register double-buffered B operands.
// ---------------------------------------------------------------------------
template <bool RELU, bool OUT_BF16>
__global__ __launch_bounds__(256) void gemm_kernel(const __bf16* __restrict__ X,
                                                   const __bf16* __restrict__ Wp,
                                                   const float* __restrict__ bias,
                                                   float* __restrict__ outf,
                                                   __bf16* __restrict__ outb) {
  __shared__ __bf16 lds_a[16 * LDS_ROW];   // 16,640 B

  const int lane = threadIdx.x & 31;
  const int wave = threadIdx.x >> 5;   // 0..7
  const int m_tile = blockIdx.x;       // 0..624
  const int l15  = lane & 15;
  const int half = lane >> 4;

  const __bf16* apanel = X + (size_t)m_tile * 16 * D;

#if HAVE_TDM
  if (threadIdx.x < 32) {              // one wave issues the TDM op
    tdm_load_a_panel(apanel, (unsigned)(uintptr_t)&lds_a[0]);
    __builtin_amdgcn_s_wait_tensorcnt(0);
  }
#else
  // cooperative fallback: 1024 chunks of 8 bf16
  for (int c = threadIdx.x; c < 1024; c += 256) {
    int row = c >> 6, col8 = (c & 63) * 8;
    *(v8bf*)&lds_a[row * LDS_ROW + col8] = *(const v8bf*)(apanel + (size_t)row * D + col8);
  }
#endif
  __syncthreads();

  // A operand from LDS (contiguous k order, as in Wpack: lane<16 -> K {k0..k0+15},
  // lane>=16 -> K {k0+16..k0+31}). Two ds_load_b128 per k-step.
  const __bf16* arow = lds_a + l15 * LDS_ROW + half * 16;

  const int nt0 = wave * 4;
  const size_t NSTEP = (size_t)16 * 32 * 16;  // next n_tile in Wpack
  const size_t KSTEP = (size_t)32 * 16;       // next k_tile in Wpack
  const __bf16* bp0 = Wp + (((size_t)nt0 * 16) * 32 + lane) * 16;

  v8f acc[4] = {v8f{}, v8f{}, v8f{}, v8f{}};

  v16bf bcur[4];
#pragma unroll
  for (int t = 0; t < 4; ++t) bcur[t] = ld16g(bp0 + t * NSTEP);

#pragma unroll
  for (int kt = 0; kt < 16; ++kt) {
    v8bf alo = *(const v8bf*)(arow + kt * 32);
    v8bf ahi = *(const v8bf*)(arow + kt * 32 + 8);
    v16bf a = __builtin_shufflevector(alo, ahi, 0, 1, 2, 3, 4, 5, 6, 7,
                                      8, 9, 10, 11, 12, 13, 14, 15);
    v16bf bnxt[4];
    if (kt < 15) {
#pragma unroll
      for (int t = 0; t < 4; ++t)
        bnxt[t] = ld16g(bp0 + (kt + 1) * KSTEP + t * NSTEP);
    }
#pragma unroll
    for (int t = 0; t < 4; ++t)
      acc[t] = __builtin_amdgcn_wmma_f32_16x16x32_bf16(
          /*neg_a=*/false, a, /*neg_b=*/false, bcur[t],
          /*c_mod=*/(short)0, acc[t], /*reuse_a=*/false, /*reuse_b=*/false);
    if (kt < 15) {
#pragma unroll
      for (int t = 0; t < 4; ++t) bcur[t] = bnxt[t];
    }
  }

  // Epilogue. C/D layout: VGPR r -> (lane<16: M=r, N=lane) (lane>=16: M=r+8, N=lane-16)
  const int rowbase = m_tile * 16 + half * 8;
#pragma unroll
  for (int t = 0; t < 4; ++t) {
    int n = (nt0 + t) * 16 + l15;
    float bv = bias[n];
#pragma unroll
    for (int r = 0; r < 8; ++r) {
      float v = acc[t][r] + bv;
      if (RELU) v = v > 0.f ? v : 0.f;
      size_t idx = (size_t)(rowbase + r) * D + n;
      if (OUT_BF16) outb[idx] = (__bf16)v;
      else          outf[idx] = v;
    }
  }
}

// ---------------------------------------------------------------------------
extern "C" void kernel_launch(void* const* d_in, const int* in_sizes, int n_in,
                              void* d_out, int out_size, void* d_ws, size_t ws_size,
                              hipStream_t stream) {
  const float*     h  = (const float*)d_in[0];
  const long long* ei = (const long long*)d_in[1];
  const float*     W1 = (const float*)d_in[2];
  const float*     b1 = (const float*)d_in[3];
  const float*     W2 = (const float*)d_in[4];
  const float*     b2 = (const float*)d_in[5];
  float* out = (float*)d_out;

  char* ws = (char*)d_ws;
  float*  agg = (float*)(ws);                    // 10000*512*4 = 20,480,000 B
  __bf16* xbf = (__bf16*)(ws + 20480000);        // 10,240,000 B
  __bf16* ybf = (__bf16*)(ws + 30720000);        // 10,240,000 B
  __bf16* wp1 = (__bf16*)(ws + 40960000);        //    524,288 B
  __bf16* wp2 = (__bf16*)(ws + 41484288);        //    524,288 B  (total ~42 MB)

  init_agg_kernel<<<(N_NODES * D / 4 + 255) / 256, 256, 0, stream>>>(
      (const float4*)h, (float4*)agg, N_NODES * D / 4);
  scatter_kernel<<<(N_EDGES * 128 + 255) / 256, 256, 0, stream>>>(h, ei, agg);
  cvt_kernel<<<(N_NODES * D / 4 + 255) / 256, 256, 0, stream>>>(agg, xbf, N_NODES * D);
  packW_kernel<<<64, 256, 0, stream>>>(W1, wp1);
  packW_kernel<<<64, 256, 0, stream>>>(W2, wp2);

  // GEMM1: y = relu(x @ W1 + b1)  -> bf16
  gemm_kernel<true, true><<<N_NODES / 16, 256, 0, stream>>>(xbf, wp1, b1, nullptr, ybf);
  // GEMM2: out = y @ W2 + b2      -> fp32
  gemm_kernel<false, false><<<N_NODES / 16, 256, 0, stream>>>(ybf, wp2, b2, out, nullptr);
}
